// FasterRCNN_48713519072066
// MI455X (gfx1250) — matrix operand order repaired and checked by reading, not compiled, not run on hardware
//
#include <hip/hip_runtime.h>
#include <math.h>

typedef _Float16 half_t;
typedef __attribute__((ext_vector_type(8)))  _Float16 v8h;
typedef __attribute__((ext_vector_type(16))) _Float16 v16h;
typedef __attribute__((ext_vector_type(8)))  float    v8f;

#define C_IN   256
#define KTOT   2304          // 256*9
#define NSTEP  72            // KTOT/32
#define TOTA   159882        // total anchors
#define KCAT   4507          // 1000+1000+1000+1000+507
#define IMGF   800.0f
#define NMS_T  0.7f
#define BBOX_CLIP 4.135166556742356f   // log(1000/16)

__constant__ int d_lvlHW[5]    = {40000, 10000, 2500, 625, 169};
__constant__ int d_lvlBaseA[5] = {0, 120000, 150000, 157500, 159375};
__constant__ int d_lvlK[5]     = {1000, 1000, 1000, 1000, 507};
__constant__ int d_lvlOff[5]   = {0, 1000, 2000, 3000, 4000};

// ---------------------------------------------------------------------------
// 1) conv weights f32 -> f16, reordered K: dst[co][tap*256+ci] = src[co][ci*9+tap]
//    (tap-major K ordering makes each 32-wide K chunk single-tap)
// ---------------------------------------------------------------------------
__global__ void repack_weights(const float* __restrict__ w, half_t* __restrict__ wf, int n) {
    int i = blockIdx.x * blockDim.x + threadIdx.x;
    if (i < n) {
        int co  = i / KTOT;
        int r   = i - co * KTOT;
        int tap = r >> 8;      // 0..8
        int ci  = r & 255;
        wf[i] = (half_t)w[(size_t)co * KTOT + ci * 9 + tap];
    }
}

// ---------------------------------------------------------------------------
// 2) 3x3 conv 256->256 + bias + ReLU as implicit GEMM via WMMA f16->f32.
//    Block: 256 threads = 8 waves; macro tile 64 co x 64 px;
//    each wave: one 16-row co slice x two 16-px tiles -> 2 WMMAs / K-step.
//    K = tap*256 + ci, stepped by 32 -> tap uniform per step (SALU).
//    Software pipeline: double-buffered LDS + register staging; global loads
//    for step i+1 overlap the WMMAs of step i; one barrier per K-step.
//    LDS:  lA [buf][co][kk] row-major (A frag = 2 contiguous 16B chunks)
//          lB [buf][px][kk] px-major  (B frag = 1 contiguous 32B chunk)
//    Output channels-last f16: t[n][pixel][co]
// ---------------------------------------------------------------------------
__global__ __launch_bounds__(256)
void conv3x3_wmma(const float* __restrict__ feat,   // [4,256,H,W]
                  const half_t* __restrict__ wf,    // [256,2304] tap-major K
                  const float* __restrict__ bias,   // [256]
                  half_t* __restrict__ t,           // [4,HW,256]
                  int H, int W)
{
    __shared__ half_t lA[2][64 * 32];   // [co_local][kk], row stride 32 (64B)
    __shared__ half_t lB[2][64 * 32];   // [px_local][kk], row stride 32 (64B)

    const int HW     = H * W;
    const int n      = blockIdx.z;
    const int coBase = blockIdx.y * 64;
    const int pxBase = blockIdx.x * 64;
    const int tid    = threadIdx.x;
    const int lane   = tid & 31;
    const int wave   = tid >> 5;
    const int coTile = wave & 3;     // 0..3 -> 16-row co slice of 64
    const int pxHalf = wave >> 2;    // 0..1 -> 32-px slice of 64
    const int mrow   = lane & 15;
    const int hi     = lane >> 4;    // lane half (K interleave select)

    const float* featN = feat + (size_t)n * C_IN * HW;

    // B staging assignment: thread -> (ci index kk=0..31, 8 consecutive px)
    const int sKk  = tid >> 3;          // 0..31
    const int sPxg = (tid & 7) * 8;     // 0,8,...,56
    const int p0   = pxBase + sPxg;
    const int y0   = p0 / W;            // hoisted out of the K loop
    const int x0   = p0 - y0 * W;

    // A staging assignment: thread -> 8 contiguous halves of [64x32]
    const half_t* aSrc = wf + (size_t)(coBase + (tid >> 2)) * KTOT + (tid & 3) * 8;

    // fetch one K-step (32-wide) into registers
    auto loadStage = [&](int step, v8h& aR, v8h& bR) {
        const int k0    = step * 32;
        const int tap   = k0 >> 8;
        const int ci0   = k0 & 255;
        const int ty    = tap / 3;
        const int dy    = ty - 1;
        const int dx    = tap - ty * 3 - 1;
        const int shift = dy * W + dx;

        aR = *(const v8h*)(aSrc + k0);
        if (k0 + 32 < KTOT)
            __builtin_prefetch(aSrc + k0 + 32, 0, 0);

        const float* src = featN + (size_t)(ci0 + sKk) * HW;
        #pragma unroll
        for (int j = 0; j < 8; ++j) {
            int xj = x0 + j;
            int yj = y0;
            if (xj >= W) { xj -= W; yj += 1; }   // select, W >= 13 > 8
            int xx = xj + dx;
            int yy = yj + dy;
            bool valid = ((p0 + j) < HW) & (yy >= 0) & (yy < H) &
                         (xx >= 0) & (xx < W);
            int addr = valid ? (p0 + j + shift) : 0;
            float v = src[addr];
            bR[j] = (half_t)(valid ? v : 0.0f);
        }
    };
    // deposit register stage into LDS buffer
    auto storeStage = [&](int buf, const v8h& aR, const v8h& bR) {
        *(v8h*)(&lA[buf][tid * 8]) = aR;
        #pragma unroll
        for (int j = 0; j < 8; ++j)
            lB[buf][(sPxg + j) * 32 + sKk] = bR[j];
    };

    v8f acc0 = {}, acc1 = {};
    v8h aR, bR, aRn, bRn;

    loadStage(0, aR, bR);
    storeStage(0, aR, bR);

    for (int step = 0; step < NSTEP; ++step) {
        __syncthreads();   // buf stores visible; prior reads of buf^1 retired
        const int buf = step & 1;

        if (step + 1 < NSTEP)
            loadStage(step + 1, aRn, bRn);   // global loads overlap WMMAs

        // ---- A fragment: two contiguous 16B chunks of one LDS row
        const half_t* arow = &lA[buf][(coTile * 16 + mrow) * 32];
        v8h a0 = *(const v8h*)(arow + 16 * hi);       // K = 16*hi .. +7
        v8h a1 = *(const v8h*)(arow + 16 * hi + 8);   // K = 16*hi+8 .. +7
        v16h a = __builtin_shufflevector(a0, a1,
                 0, 1, 2, 3, 4, 5, 6, 7, 8, 9, 10, 11, 12, 13, 14, 15);

        // ---- B fragments: one contiguous 32B chunk per column
        v16h b0 = *(const v16h*)(&lB[buf][(pxHalf * 32 + mrow) * 32 + 16 * hi]);
        v16h b1 = *(const v16h*)(&lB[buf][(pxHalf * 32 + 16 + mrow) * 32 + 16 * hi]);

        acc0 = __builtin_amdgcn_wmma_f32_16x16x32_f16(
            false, a, false, b0, (short)0, acc0, false, false);
        acc1 = __builtin_amdgcn_wmma_f32_16x16x32_f16(
            false, a, false, b1, (short)0, acc1, false, false);

        if (step + 1 < NSTEP)
            storeStage(buf ^ 1, aRn, bRn);   // fill other buffer for next step
    }

    // ---- epilogue: bias + ReLU; 8 accumulator rows = 8 consecutive co
    const int coOut = coBase + coTile * 16 + hi * 8;
    #pragma unroll
    for (int sub = 0; sub < 2; ++sub) {
        const v8f& acc = sub ? acc1 : acc0;
        int p = pxBase + pxHalf * 32 + sub * 16 + mrow;
        v8h o;
        #pragma unroll
        for (int r = 0; r < 8; ++r) {
            float v = acc[r] + bias[coOut + r];
            o[r] = (half_t)(v > 0.0f ? v : 0.0f);
        }
        if (p < HW)
            *(v8h*)(&t[((size_t)n * HW + p) * C_IN + coOut]) = o;
    }
}

// ---------------------------------------------------------------------------
// 3) 1x1 heads: per (pixel, output o in 0..14) dot over 256 channels.
//    o<3 -> cls logit (anchor o); o>=3 -> bbox reg channel (o-3)=a*4+c.
// ---------------------------------------------------------------------------
__global__ void head1x1(const half_t* __restrict__ tL,   // [4,HW,256]
                        const float* __restrict__ clsW, const float* __restrict__ clsB,
                        const float* __restrict__ bbW,  const float* __restrict__ bbB,
                        float* __restrict__ obj,         // [4,TOTA]
                        float* __restrict__ reg,         // [4,TOTA,4]
                        int HW, int anchorBase)
{
    long long idx = (long long)blockIdx.x * blockDim.x + threadIdx.x;
    int n = blockIdx.y;
    if (idx >= (long long)HW * 15) return;
    int p = (int)(idx / 15);
    int o = (int)(idx % 15);

    const v8h* row = (const v8h*)(tL + ((size_t)n * HW + p) * C_IN);
    const float* wrow;
    float s;
    if (o < 3) { wrow = clsW + o * C_IN;       s = clsB[o]; }
    else       { wrow = bbW + (o - 3) * C_IN;  s = bbB[o - 3]; }
    #pragma unroll 4
    for (int k = 0; k < C_IN / 8; ++k) {
        v8h h = row[k];
        #pragma unroll
        for (int j = 0; j < 8; ++j) s += (float)h[j] * wrow[k * 8 + j];
    }

    if (o < 3) {
        int gi = anchorBase + p * 3 + o;
        obj[(size_t)n * TOTA + gi] = s;
    } else {
        int a  = (o - 3) >> 2;
        int c  = (o - 3) & 3;
        int gi = anchorBase + p * 3 + a;
        reg[((size_t)n * TOTA + gi) * 4 + c] = s;
    }
}

// ---------------------------------------------------------------------------
// 4) Box decode + clip to [0, 800]
// ---------------------------------------------------------------------------
__global__ void decode_boxes(const float* __restrict__ anchors,
                             const float* __restrict__ reg,
                             float* __restrict__ boxes)
{
    long long i = (long long)blockIdx.x * blockDim.x + threadIdx.x;
    if (i >= 4LL * TOTA) return;
    int a = (int)(i % TOTA);

    float x1 = anchors[a * 4 + 0], y1 = anchors[a * 4 + 1];
    float x2 = anchors[a * 4 + 2], y2 = anchors[a * 4 + 3];
    float aw = x2 - x1, ah = y2 - y1;
    float ax = x1 + 0.5f * aw, ay = y1 + 0.5f * ah;

    const float* r = reg + (size_t)i * 4;
    float dx = r[0], dy = r[1];
    float dw = fminf(r[2], BBOX_CLIP);
    float dh = fminf(r[3], BBOX_CLIP);
    float px = dx * aw + ax, py = dy * ah + ay;
    float pw = expf(dw) * aw, ph = expf(dh) * ah;

    float bx1 = fminf(fmaxf(px - 0.5f * pw, 0.0f), IMGF);
    float by1 = fminf(fmaxf(py - 0.5f * ph, 0.0f), IMGF);
    float bx2 = fminf(fmaxf(px + 0.5f * pw, 0.0f), IMGF);
    float by2 = fminf(fmaxf(py + 0.5f * ph, 0.0f), IMGF);

    float* b = boxes + (size_t)i * 4;
    b[0] = bx1; b[1] = by1; b[2] = bx2; b[3] = by2;
}

// ---------------------------------------------------------------------------
// 5) Per-level top-k by rank counting (stable ties: earlier index wins)
// ---------------------------------------------------------------------------
__global__ void topk_rank(const float* __restrict__ obj, int* __restrict__ selIdx)
{
    int l = blockIdx.x, n = blockIdx.y;
    int base = d_lvlBaseA[l];
    int cnt  = 3 * d_lvlHW[l];
    int K    = d_lvlK[l];
    int off  = d_lvlOff[l];
    const float* s = obj + (size_t)n * TOTA + base;
    for (int i = threadIdx.x; i < cnt; i += blockDim.x) {
        float si = s[i];
        int rank = 0;
        for (int j = 0; j < cnt; ++j) {
            float sj = s[j];
            rank += (sj > si) || (sj == si && j < i);
        }
        if (rank < K) selIdx[(size_t)n * KCAT + off + rank] = base + i;
    }
}

// ---------------------------------------------------------------------------
// 6) Gather: sigmoid scores, validity mask (-inf), level-offset NMS boxes
// ---------------------------------------------------------------------------
__global__ void gather_cands(const float* __restrict__ obj,
                             const float* __restrict__ boxes,
                             const int*   __restrict__ selIdx,
                             float* __restrict__ cScore,
                             float* __restrict__ cBox,
                             float* __restrict__ cNms)
{
    int j = blockIdx.x * blockDim.x + threadIdx.x;
    int n = blockIdx.y;
    if (j >= KCAT) return;
    int l = j / 1000; if (l > 4) l = 4;
    int i = selIdx[(size_t)n * KCAT + j];

    float s = 1.0f / (1.0f + expf(-obj[(size_t)n * TOTA + i]));
    const float* b = boxes + ((size_t)n * TOTA + i) * 4;
    float x1 = b[0], y1 = b[1], x2 = b[2], y2 = b[3];
    bool valid = ((x2 - x1) >= 1e-3f) && ((y2 - y1) >= 1e-3f);
    if (!valid) s = -INFINITY;

    size_t o = (size_t)n * KCAT + j;
    cScore[o] = s;
    float* cb = cBox + o * 4;
    cb[0] = x1; cb[1] = y1; cb[2] = x2; cb[3] = y2;
    float off = (float)l * (IMGF + 1.0f);
    float* cn = cNms + o * 4;
    cn[0] = x1 + off; cn[1] = y1 + off; cn[2] = x2 + off; cn[3] = y2 + off;
}

// ---------------------------------------------------------------------------
// 7) Global stable sort (rank counting over KCAT=4507)
// ---------------------------------------------------------------------------
__global__ void sort_cands(const float* __restrict__ cScore,
                           const float* __restrict__ cBox,
                           const float* __restrict__ cNms,
                           float* __restrict__ sScore,
                           float* __restrict__ sBox,
                           float* __restrict__ sNms)
{
    int i = blockIdx.x * blockDim.x + threadIdx.x;
    int n = blockIdx.y;
    if (i >= KCAT) return;
    const float* s = cScore + (size_t)n * KCAT;
    float si = s[i];
    int rank = 0;
    for (int j = 0; j < KCAT; ++j) {
        float sj = s[j];
        rank += (sj > si) || (sj == si && j < i);
    }
    size_t src = (size_t)n * KCAT + i;
    size_t dst = (size_t)n * KCAT + rank;
    sScore[dst] = si;
    for (int c = 0; c < 4; ++c) {
        sBox[dst * 4 + c] = cBox[src * 4 + c];
        sNms[dst * 4 + c] = cNms[src * 4 + c];
    }
}

// ---------------------------------------------------------------------------
// 8) NMS: sequential over i, parallel suppression over j (one block / image)
// ---------------------------------------------------------------------------
__global__ void nms_kernel(const float* __restrict__ sNms, int* __restrict__ kept)
{
    int n = blockIdx.x;
    const float* b = sNms + (size_t)n * KCAT * 4;
    int* kp = kept + (size_t)n * KCAT;
    for (int j = threadIdx.x; j < KCAT; j += blockDim.x) kp[j] = 1;
    __syncthreads();
    for (int i = 0; i < KCAT; ++i) {
        if (kp[i]) {
            float ix1 = b[i * 4 + 0], iy1 = b[i * 4 + 1];
            float ix2 = b[i * 4 + 2], iy2 = b[i * 4 + 3];
            float ia  = (ix2 - ix1) * (iy2 - iy1);
            for (int j = threadIdx.x; j < KCAT; j += blockDim.x) {
                if (j > i && kp[j]) {
                    float xx1 = fmaxf(ix1, b[j * 4 + 0]);
                    float yy1 = fmaxf(iy1, b[j * 4 + 1]);
                    float xx2 = fminf(ix2, b[j * 4 + 2]);
                    float yy2 = fminf(iy2, b[j * 4 + 3]);
                    float w = fmaxf(xx2 - xx1, 0.0f);
                    float h = fmaxf(yy2 - yy1, 0.0f);
                    float inter = w * h;
                    float ja = (b[j * 4 + 2] - b[j * 4 + 0]) * (b[j * 4 + 3] - b[j * 4 + 1]);
                    float iou = inter / (ia + ja - inter);
                    if (iou > NMS_T) kp[j] = 0;
                }
            }
        }
        __syncthreads();
    }
}

// ---------------------------------------------------------------------------
// 9) Emit top POST_NMS kept+finite candidates, zero-padded
// ---------------------------------------------------------------------------
__global__ void finalize(const float* __restrict__ sScore,
                         const float* __restrict__ sBox,
                         const int*   __restrict__ kept,
                         float* __restrict__ out)
{
    int n = blockIdx.x;
    if (threadIdx.x != 0) return;
    float* ob = out + (size_t)n * 1000 * 4;
    float* os = out + 16000 + (size_t)n * 1000;
    int cnt = 0;
    for (int r = 0; r < KCAT && cnt < 1000; ++r) {
        size_t o = (size_t)n * KCAT + r;
        float s = sScore[o];
        if (kept[o] && s > -1e30f) {
            ob[cnt * 4 + 0] = sBox[o * 4 + 0];
            ob[cnt * 4 + 1] = sBox[o * 4 + 1];
            ob[cnt * 4 + 2] = sBox[o * 4 + 2];
            ob[cnt * 4 + 3] = sBox[o * 4 + 3];
            os[cnt] = s;
            ++cnt;
        }
    }
    for (; cnt < 1000; ++cnt) {
        ob[cnt * 4 + 0] = 0.0f; ob[cnt * 4 + 1] = 0.0f;
        ob[cnt * 4 + 2] = 0.0f; ob[cnt * 4 + 3] = 0.0f;
        os[cnt] = 0.0f;
    }
}

// ---------------------------------------------------------------------------
extern "C" void kernel_launch(void* const* d_in, const int* in_sizes, int n_in,
                              void* d_out, int out_size, void* d_ws, size_t ws_size,
                              hipStream_t stream) {
    const float* feats[5] = {(const float*)d_in[0], (const float*)d_in[1],
                             (const float*)d_in[2], (const float*)d_in[3],
                             (const float*)d_in[4]};
    const float* conv_w  = (const float*)d_in[5];
    const float* conv_b  = (const float*)d_in[6];
    const float* cls_w   = (const float*)d_in[7];
    const float* cls_b   = (const float*)d_in[8];
    const float* bbox_w  = (const float*)d_in[9];
    const float* bbox_b  = (const float*)d_in[10];
    const float* anchors = (const float*)d_in[11];
    float* out = (float*)d_out;

    static const int Hs[5]     = {200, 100, 50, 25, 13};
    static const int Ws[5]     = {200, 100, 50, 25, 13};
    static const int HWs[5]    = {40000, 10000, 2500, 625, 169};
    static const int cumHW[5]  = {0, 40000, 50000, 52500, 53125};
    static const int baseA[5]  = {0, 120000, 150000, 157500, 159375};
    const int totHW = 53294;

    // workspace carve-up
    char* p = (char*)d_ws;
    auto carve = [&](size_t bytes) {
        void* r = (void*)p;
        p += (bytes + 255) & ~(size_t)255;
        return r;
    };
    half_t* wf16   = (half_t*)carve((size_t)256 * KTOT * sizeof(half_t));
    half_t* tbuf   = (half_t*)carve((size_t)4 * totHW * C_IN * sizeof(half_t));
    float*  obj    = (float*)carve((size_t)4 * TOTA * sizeof(float));
    float*  reg    = (float*)carve((size_t)4 * TOTA * 4 * sizeof(float));
    float*  boxes  = (float*)carve((size_t)4 * TOTA * 4 * sizeof(float));
    int*    selIdx = (int*)carve((size_t)4 * KCAT * sizeof(int));
    float*  cScore = (float*)carve((size_t)4 * KCAT * sizeof(float));
    float*  cBox   = (float*)carve((size_t)4 * KCAT * 4 * sizeof(float));
    float*  cNms   = (float*)carve((size_t)4 * KCAT * 4 * sizeof(float));
    float*  sScore = (float*)carve((size_t)4 * KCAT * sizeof(float));
    float*  sBox   = (float*)carve((size_t)4 * KCAT * 4 * sizeof(float));
    float*  sNms   = (float*)carve((size_t)4 * KCAT * 4 * sizeof(float));
    int*    kept   = (int*)carve((size_t)4 * KCAT * sizeof(int));

    // 1) weight repack f32->f16 with tap-major K reorder
    {
        int nw = 256 * KTOT;
        repack_weights<<<(nw + 255) / 256, 256, 0, stream>>>(conv_w, wf16, nw);
    }

    // 2) WMMA conv + 3) heads per level
    for (int l = 0; l < 5; ++l) {
        int H = Hs[l], W = Ws[l], HW = HWs[l];
        half_t* tL = tbuf + (size_t)4 * cumHW[l] * C_IN;
        dim3 grid((HW + 63) / 64, 4, 4);
        conv3x3_wmma<<<grid, 256, 0, stream>>>(feats[l], wf16, conv_b, tL, H, W);

        long long work = (long long)HW * 15;
        dim3 hgrid((unsigned)((work + 255) / 256), 4, 1);
        head1x1<<<hgrid, 256, 0, stream>>>(tL, cls_w, cls_b, bbox_w, bbox_b,
                                           obj, reg, HW, baseA[l]);
    }

    // 4) decode
    {
        long long n = 4LL * TOTA;
        decode_boxes<<<(unsigned)((n + 255) / 256), 256, 0, stream>>>(anchors, reg, boxes);
    }

    // 5) per-level top-k
    topk_rank<<<dim3(5, 4, 1), 256, 0, stream>>>(obj, selIdx);

    // 6) gather candidates
    gather_cands<<<dim3((KCAT + 255) / 256, 4, 1), 256, 0, stream>>>(
        obj, boxes, selIdx, cScore, cBox, cNms);

    // 7) stable sort by score
    sort_cands<<<dim3((KCAT + 255) / 256, 4, 1), 256, 0, stream>>>(
        cScore, cBox, cNms, sScore, sBox, sNms);

    // 8) NMS
    nms_kernel<<<4, 256, 0, stream>>>(sNms, kept);

    // 9) finalize
    finalize<<<4, 32, 0, stream>>>(sScore, sBox, kept, out);
}
